// SimpleMoE_14577119003373
// MI455X (gfx1250) — matrix operand order repaired and checked by reading, not compiled, not run on hardware
//
#include <hip/hip_runtime.h>
#include <hip/hip_bf16.h>

// Problem constants (match reference)
constexpr int Bb = 4, Tt = 4096, Hh = 2048, Ee = 8, Dd = 1024;
constexpr int BT = Bb * Tt;          // 16384 tokens
constexpr int LDK = 40;              // LDS row stride (32 + 8 pad), 80B: 16B-aligned rows

typedef __bf16 bf16_t;
typedef bf16_t v16bf __attribute__((ext_vector_type(16)));
typedef unsigned short v16us __attribute__((ext_vector_type(16)));
typedef float v8f __attribute__((ext_vector_type(8)));

union V16 { v16us u; v16bf b; };

// fp32 pair -> packed bf16 pair, round-half-up.
// 2x v_add + 1x v_perm_b32 (merge of the two high halves in one op).
static __device__ __forceinline__ unsigned pack2(float a, float b) {
    const unsigned ua = __float_as_uint(a) + 0x8000u;
    const unsigned ub = __float_as_uint(b) + 0x8000u;
#if __has_builtin(__builtin_amdgcn_perm)
    return __builtin_amdgcn_perm(ub, ua, 0x07060302u); // {ub[3:2], ua[3:2]}
#else
    return (ua >> 16) | (ub & 0xFFFF0000u);
#endif
}

// A fragment (16x32 bf16) from LDS tile stored [M][K] (stride LDK).
static __device__ __forceinline__ v16bf load_a_frag(const unsigned short* As, int mbase, int lane) {
    const unsigned short* p = As + (mbase + (lane & 15)) * LDK + ((lane >> 4) << 3);
    V16 r;
#pragma unroll
    for (int j = 0; j < 8; ++j) { r.u[j] = p[j]; r.u[j + 8] = p[j + 16]; }
    return r.b;
}
// B fragment (32x16 bf16) from LDS tile stored TRANSPOSED [N][K] (stride LDK).
static __device__ __forceinline__ v16bf load_b_frag(const unsigned short* Bs, int nbase, int lane) {
    const unsigned short* p = Bs + (nbase + (lane & 15)) * LDK + ((lane >> 4) << 3);
    V16 r;
#pragma unroll
    for (int j = 0; j < 8; ++j) { r.u[j] = p[j]; r.u[j + 8] = p[j + 16]; }
    return r.b;
}

// ---------------------------------------------------------------------------
// Kernel 0: zero expert counters
// ---------------------------------------------------------------------------
__global__ void moe_init(int* __restrict__ cnt) {
    if (threadIdx.x < Ee) cnt[threadIdx.x] = 0;
}

// ---------------------------------------------------------------------------
// Kernel 1: router logits -> softmax -> top-2 -> per-expert token buckets
// ---------------------------------------------------------------------------
__global__ __launch_bounds__(256) void moe_router(
    const float* __restrict__ x, const float* __restrict__ Wr, const float* __restrict__ br,
    int* __restrict__ cnt, int* __restrict__ btok, float* __restrict__ bgate,
    int* __restrict__ bslot, int* __restrict__ t2i, float* __restrict__ t2s)
{
    const int lane = threadIdx.x & 31;
    const int wave = threadIdx.x >> 5;
    const int t = blockIdx.x * 8 + wave;
    if (t >= BT) return;

    float acc[Ee];
#pragma unroll
    for (int e = 0; e < Ee; ++e) acc[e] = 0.0f;

    const float4* xr4 = reinterpret_cast<const float4*>(x + (size_t)t * Hh);
    for (int h4 = lane; h4 < Hh / 4; h4 += 32) {
        const float4 xv = xr4[h4];
        const float* w = Wr + (size_t)h4 * 4 * Ee;
#pragma unroll
        for (int e = 0; e < Ee; ++e)
            acc[e] += xv.x * w[e] + xv.y * w[Ee + e] + xv.z * w[2 * Ee + e] + xv.w * w[3 * Ee + e];
    }
#pragma unroll
    for (int e = 0; e < Ee; ++e) {
#pragma unroll
        for (int off = 16; off > 0; off >>= 1) acc[e] += __shfl_xor(acc[e], off, 32);
    }

    if (lane == 0) {
        float lg[Ee], p[Ee];
        float mx = -3.4e38f;
#pragma unroll
        for (int e = 0; e < Ee; ++e) { lg[e] = acc[e] + br[e]; mx = fmaxf(mx, lg[e]); }
        float den = 0.0f;
#pragma unroll
        for (int e = 0; e < Ee; ++e) { p[e] = __expf(lg[e] - mx); den += p[e]; }
        const float inv = 1.0f / den;
#pragma unroll
        for (int e = 0; e < Ee; ++e) p[e] *= inv;

        int e0 = 0; float s0 = p[0];
#pragma unroll
        for (int e = 1; e < Ee; ++e) if (p[e] > s0) { s0 = p[e]; e0 = e; }
        int e1 = -1; float s1 = -3.4e38f;
#pragma unroll
        for (int e = 0; e < Ee; ++e) if (e != e0 && p[e] > s1) { s1 = p[e]; e1 = e; }

        t2i[t * 2 + 0] = e0; t2i[t * 2 + 1] = e1;
        t2s[t * 2 + 0] = s0; t2s[t * 2 + 1] = s1;

        const int   es[2] = { e0, e1 };
        const float ss[2] = { s0, s1 };
#pragma unroll
        for (int k = 0; k < 2; ++k) {
            const int pos = atomicAdd(&cnt[es[k]], 1);
            const int b = es[k] * BT + pos;
            btok[b] = t; bgate[b] = ss[k]; bslot[b] = k;
        }
    }
}

// ---------------------------------------------------------------------------
// Kernel 2: grouped (gather) expert GEMM, bf16 WMMA, fp32 accumulate.
// 128x128x32 block tile, 8 wave32s (each 64x32 = 4x2 WMMA accumulators),
// double-buffered LDS software pipeline: prefetch -> WMMA -> commit -> barrier.
// ---------------------------------------------------------------------------
__global__ __launch_bounds__(256) void moe_expert_gemm(
    const float* __restrict__ x, const float* __restrict__ We,
    const int* __restrict__ cnt, const int* __restrict__ btok,
    const float* __restrict__ bgate, const int* __restrict__ bslot,
    float* __restrict__ partial)
{
    __shared__ alignas(16) unsigned short As[2][128 * LDK];
    __shared__ alignas(16) unsigned short Bs[2][128 * LDK];
    __shared__ int   s_tok[128];
    __shared__ float s_gate[128];
    __shared__ int   s_slot[128];

    constexpr int tilesPerExp = BT / 128;
    const int e = blockIdx.x / tilesPerExp;
    const int mt = blockIdx.x % tilesPerExp;
    const int rows = cnt[e];
    const int row0 = mt * 128;
    if (row0 >= rows) return;
    const int nrows = min(128, rows - row0);
    const int n0 = blockIdx.y * 128;

    const int tid = threadIdx.x;
    if (tid < 128) {
        const bool v = tid < nrows;
        const int b = e * BT + row0 + tid;
        s_tok[tid]  = v ? btok[b]  : 0;   // invalid rows read token 0 (discarded at store)
        s_gate[tid] = v ? bgate[b] : 0.0f;
        s_slot[tid] = v ? bslot[b] : 0;
    }
    __syncthreads();

    const int lane = tid & 31, wave = tid >> 5;
    const int wm = (wave & 1) * 64;
    const int wn = (wave >> 1) * 32;

    const v8f vzero = { 0, 0, 0, 0, 0, 0, 0, 0 };
    v8f acc[4][2];
#pragma unroll
    for (int i = 0; i < 4; ++i)
#pragma unroll
        for (int j = 0; j < 2; ++j) acc[i][j] = vzero;

    // A staging: thread covers rows arow+32i (i=0..3), 4 consecutive k each.
    const int arow  = tid >> 3;          // 0..31
    const int acol  = (tid & 7) * 4;     // 0,4,..,28
    const float* aptr[4];
#pragma unroll
    for (int i = 0; i < 4; ++i)
        aptr[i] = x + (size_t)s_tok[arow + 32 * i] * Hh + acol;

    // B staging: thread covers column bn, k = bk..bk+15 (coalesced across lanes).
    const int bn = tid & 127;
    const int bk = (tid >> 7) * 16;      // 0 or 16
    const float* Wep = We + (size_t)e * Hh * Dd;
    const float* bptr = Wep + (size_t)bk * Dd + n0 + bn;

    float4 apre[4];
    float  bpre[16];

    auto prefetch = [&](int kk) {
#pragma unroll
        for (int i = 0; i < 4; ++i) apre[i] = *reinterpret_cast<const float4*>(aptr[i] + kk);
        const float* bp = bptr + (size_t)kk * Dd;
#pragma unroll
        for (int j = 0; j < 16; ++j) bpre[j] = bp[(size_t)j * Dd];
    };
    auto commit = [&](int buf) {
        unsigned short* Asb = As[buf];
        unsigned short* Bsb = Bs[buf];
#pragma unroll
        for (int i = 0; i < 4; ++i) {
            uint2 pk;
            pk.x = pack2(apre[i].x, apre[i].y);
            pk.y = pack2(apre[i].z, apre[i].w);
            *reinterpret_cast<uint2*>(&Asb[(arow + 32 * i) * LDK + acol]) = pk;
        }
        uint4 q0, q1;
        q0.x = pack2(bpre[0],  bpre[1]);  q0.y = pack2(bpre[2],  bpre[3]);
        q0.z = pack2(bpre[4],  bpre[5]);  q0.w = pack2(bpre[6],  bpre[7]);
        q1.x = pack2(bpre[8],  bpre[9]);  q1.y = pack2(bpre[10], bpre[11]);
        q1.z = pack2(bpre[12], bpre[13]); q1.w = pack2(bpre[14], bpre[15]);
        *reinterpret_cast<uint4*>(&Bsb[bn * LDK + bk])     = q0;
        *reinterpret_cast<uint4*>(&Bsb[bn * LDK + bk + 8]) = q1;
    };

    constexpr int NSTEP = Hh / 32;
    prefetch(0);
    commit(0);
    __syncthreads();

    for (int step = 0; step < NSTEP; ++step) {
        const int cur = step & 1;
        if (step + 1 < NSTEP) prefetch((step + 1) * 32);

        v16bf a[4], b[2];
#pragma unroll
        for (int ms = 0; ms < 4; ++ms) a[ms] = load_a_frag(As[cur], wm + ms * 16, lane);
#pragma unroll
        for (int ns = 0; ns < 2; ++ns) b[ns] = load_b_frag(Bs[cur], wn + ns * 16, lane);
#pragma unroll
        for (int ms = 0; ms < 4; ++ms)
#pragma unroll
            for (int ns = 0; ns < 2; ++ns)
                acc[ms][ns] = __builtin_amdgcn_wmma_f32_16x16x32_bf16(
                    false, a[ms], false, b[ns], (short)0, acc[ms][ns], false, false);

        if (step + 1 < NSTEP) commit(cur ^ 1);
        __syncthreads();
    }

    // Epilogue: scale by gate, scatter by token id into partial[slot][tok][n]
    const int cn  = lane & 15;
    const int crb = (lane >> 4) * 8;
#pragma unroll
    for (int ms = 0; ms < 4; ++ms) {
#pragma unroll
        for (int ns = 0; ns < 2; ++ns) {
            const int ncol = n0 + wn + ns * 16 + cn;
#pragma unroll
            for (int r = 0; r < 8; ++r) {
                const int row = wm + ms * 16 + crb + r;
                if (row < nrows) {
                    partial[(size_t)s_slot[row] * BT * Dd + (size_t)s_tok[row] * Dd + ncol] =
                        s_gate[row] * acc[ms][ns][r];
                }
            }
        }
    }
}

// ---------------------------------------------------------------------------
// Kernel 3: weighted = p0 + p1 + s0*be[e0] + s1*be[e1]  (in place into p0)
// ---------------------------------------------------------------------------
__global__ __launch_bounds__(256) void moe_combine(
    float* __restrict__ p0, const float* __restrict__ p1,
    const int* __restrict__ t2i, const float* __restrict__ t2s,
    const float* __restrict__ be)
{
    const size_t i4 = ((size_t)blockIdx.x * 256 + threadIdx.x) * 4;
    const int t = (int)(i4 / Dd);
    const int d = (int)(i4 % Dd);
    const int e0 = t2i[t * 2 + 0], e1 = t2i[t * 2 + 1];
    const float s0 = t2s[t * 2 + 0], s1 = t2s[t * 2 + 1];
    float4 a = *reinterpret_cast<const float4*>(p0 + i4);
    const float4 b  = *reinterpret_cast<const float4*>(p1 + i4);
    const float4 c0 = *reinterpret_cast<const float4*>(be + (size_t)e0 * Dd + d);
    const float4 c1 = *reinterpret_cast<const float4*>(be + (size_t)e1 * Dd + d);
    a.x = a.x + b.x + s0 * c0.x + s1 * c1.x;
    a.y = a.y + b.y + s0 * c0.y + s1 * c1.y;
    a.z = a.z + b.z + s0 * c0.z + s1 * c1.z;
    a.w = a.w + b.w + s0 * c0.w + s1 * c1.w;
    *reinterpret_cast<float4*>(p0 + i4) = a;
}

// ---------------------------------------------------------------------------
// Kernel 4: output GEMM  [BT,D] @ [D,H] + bc  -> out [BT,H]  (same pipeline)
// ---------------------------------------------------------------------------
__global__ __launch_bounds__(256) void moe_out_gemm(
    const float* __restrict__ A, const float* __restrict__ Wc,
    const float* __restrict__ bc, float* __restrict__ out)
{
    __shared__ alignas(16) unsigned short As[2][128 * LDK];
    __shared__ alignas(16) unsigned short Bs[2][128 * LDK];

    const int m0 = blockIdx.x * 128;
    const int n0 = blockIdx.y * 128;
    const int tid = threadIdx.x, lane = tid & 31, wave = tid >> 5;
    const int wm = (wave & 1) * 64;
    const int wn = (wave >> 1) * 32;

    const v8f vzero = { 0, 0, 0, 0, 0, 0, 0, 0 };
    v8f acc[4][2];
#pragma unroll
    for (int i = 0; i < 4; ++i)
#pragma unroll
        for (int j = 0; j < 2; ++j) acc[i][j] = vzero;

    const int arow = tid >> 3;
    const int acol = (tid & 7) * 4;
    const float* aptr[4];
#pragma unroll
    for (int i = 0; i < 4; ++i)
        aptr[i] = A + (size_t)(m0 + arow + 32 * i) * Dd + acol;

    const int bn = tid & 127;
    const int bk = (tid >> 7) * 16;
    const float* bptr = Wc + (size_t)bk * Hh + n0 + bn;

    float4 apre[4];
    float  bpre[16];

    auto prefetch = [&](int kk) {
#pragma unroll
        for (int i = 0; i < 4; ++i) apre[i] = *reinterpret_cast<const float4*>(aptr[i] + kk);
        const float* bp = bptr + (size_t)kk * Hh;
#pragma unroll
        for (int j = 0; j < 16; ++j) bpre[j] = bp[(size_t)j * Hh];
    };
    auto commit = [&](int buf) {
        unsigned short* Asb = As[buf];
        unsigned short* Bsb = Bs[buf];
#pragma unroll
        for (int i = 0; i < 4; ++i) {
            uint2 pk;
            pk.x = pack2(apre[i].x, apre[i].y);
            pk.y = pack2(apre[i].z, apre[i].w);
            *reinterpret_cast<uint2*>(&Asb[(arow + 32 * i) * LDK + acol]) = pk;
        }
        uint4 q0, q1;
        q0.x = pack2(bpre[0],  bpre[1]);  q0.y = pack2(bpre[2],  bpre[3]);
        q0.z = pack2(bpre[4],  bpre[5]);  q0.w = pack2(bpre[6],  bpre[7]);
        q1.x = pack2(bpre[8],  bpre[9]);  q1.y = pack2(bpre[10], bpre[11]);
        q1.z = pack2(bpre[12], bpre[13]); q1.w = pack2(bpre[14], bpre[15]);
        *reinterpret_cast<uint4*>(&Bsb[bn * LDK + bk])     = q0;
        *reinterpret_cast<uint4*>(&Bsb[bn * LDK + bk + 8]) = q1;
    };

    constexpr int NSTEP = Dd / 32;
    prefetch(0);
    commit(0);
    __syncthreads();

    for (int step = 0; step < NSTEP; ++step) {
        const int cur = step & 1;
        if (step + 1 < NSTEP) prefetch((step + 1) * 32);

        v16bf a[4], b[2];
#pragma unroll
        for (int ms = 0; ms < 4; ++ms) a[ms] = load_a_frag(As[cur], wm + ms * 16, lane);
#pragma unroll
        for (int ns = 0; ns < 2; ++ns) b[ns] = load_b_frag(Bs[cur], wn + ns * 16, lane);
#pragma unroll
        for (int ms = 0; ms < 4; ++ms)
#pragma unroll
            for (int ns = 0; ns < 2; ++ns)
                acc[ms][ns] = __builtin_amdgcn_wmma_f32_16x16x32_bf16(
                    false, a[ms], false, b[ns], (short)0, acc[ms][ns], false, false);

        if (step + 1 < NSTEP) commit(cur ^ 1);
        __syncthreads();
    }

    const int cn  = lane & 15;
    const int crb = (lane >> 4) * 8;
#pragma unroll
    for (int ms = 0; ms < 4; ++ms) {
#pragma unroll
        for (int ns = 0; ns < 2; ++ns) {
            const int ncol = n0 + wn + ns * 16 + cn;
            const float bias = bc[ncol];
#pragma unroll
            for (int r = 0; r < 8; ++r) {
                const int row = m0 + wm + ms * 16 + crb + r;
                out[(size_t)row * Hh + ncol] = acc[ms][ns][r] + bias;
            }
        }
    }
}

// ---------------------------------------------------------------------------
extern "C" void kernel_launch(void* const* d_in, const int* in_sizes, int n_in,
                              void* d_out, int out_size, void* d_ws, size_t ws_size,
                              hipStream_t stream) {
    (void)in_sizes; (void)n_in; (void)out_size; (void)ws_size;
    const float* x  = (const float*)d_in[0];
    const float* Wr = (const float*)d_in[1];
    const float* br = (const float*)d_in[2];
    const float* We = (const float*)d_in[3];
    const float* be = (const float*)d_in[4];
    const float* Wc = (const float*)d_in[5];
    const float* bc = (const float*)d_in[6];
    float* out = (float*)d_out;

    char* ws = (char*)d_ws;
    size_t off = 0;
    auto alloc = [&](size_t bytes) -> void* {
        void* p = ws + off;
        off += (bytes + 255) & ~(size_t)255;
        return p;
    };
    int*   cnt     = (int*)  alloc((size_t)Ee * sizeof(int));
    int*   btok    = (int*)  alloc((size_t)Ee * BT * sizeof(int));
    float* bgate   = (float*)alloc((size_t)Ee * BT * sizeof(float));
    int*   bslot   = (int*)  alloc((size_t)Ee * BT * sizeof(int));
    int*   t2i     = (int*)  alloc((size_t)BT * 2 * sizeof(int));
    float* t2s     = (float*)alloc((size_t)BT * 2 * sizeof(float));
    float* partial = (float*)alloc((size_t)2 * BT * Dd * sizeof(float));

    moe_init<<<dim3(1), dim3(32), 0, stream>>>(cnt);
    moe_router<<<dim3(BT / 8), dim3(256), 0, stream>>>(
        x, Wr, br, cnt, btok, bgate, bslot, t2i, t2s);
    moe_expert_gemm<<<dim3(Ee * (BT / 128), Dd / 128), dim3(256), 0, stream>>>(
        x, We, cnt, btok, bgate, bslot, partial);
    moe_combine<<<dim3((unsigned)(((size_t)BT * Dd) / 1024)), dim3(256), 0, stream>>>(
        partial, partial + (size_t)BT * Dd, t2i, t2s, be);
    moe_out_gemm<<<dim3(BT / 128, Hh / 128), dim3(256), 0, stream>>>(
        partial, Wc, bc, out);
}
